// LukeEntityAwareAttentionModel_12275016531992
// MI455X (gfx1250) — compile-verified, matrix operand order checked by reading, not run on hardware
//
#include <hip/hip_runtime.h>

#define LAYERS 4
#define BB 2
#define LW 1024
#define LE 64
#define S_TOT 1088
#define H 768
#define NH 12
#define DH 64
#define FF 3072
#define LDSROW 1104   // padded row stride for LDS score/prob rows (16B aligned)

typedef __bf16 v16bf __attribute__((ext_vector_type(16)));
typedef float  v8f  __attribute__((ext_vector_type(8)));
typedef unsigned int u32x4 __attribute__((ext_vector_type(4)));
typedef unsigned int u32x8 __attribute__((ext_vector_type(8)));
typedef unsigned short u16;

union Frag {
    v16bf v;
    uint4 q[2];
};

__device__ __forceinline__ u16 f32_to_bf16(float f) {
    unsigned int u = __float_as_uint(f);
    unsigned int r = u + 0x7FFFu + ((u >> 16) & 1u);
    return (u16)(r >> 16);
}

// ---------------------------------------------------------------------------
// Tensor Data Mover: async-load a 2D bf16 tile (tileK x tileN, row stride K
// elements) from global into LDS. D# built per CDNA5 ISA.
// Issued via inline asm (portable across ROCm7.2 / amdgpu-toolchain builtins).
// Completion tracked with TENSORcnt (in-order per wave).
// ---------------------------------------------------------------------------
__device__ __forceinline__ void tdm_load_tile_bf16(const u16* gsrc, unsigned ldsAddr,
                                                   int K, int tileK, int tileN) {
    unsigned long long ga = (unsigned long long)(uintptr_t)gsrc;
    u32x4 g0;
    g0[0] = 1u;                                      // count=1, user descriptor
    g0[1] = ldsAddr;                                 // lds_addr (bytes)
    g0[2] = (unsigned)(ga & 0xFFFFFFFFu);            // global_addr[31:0]
    g0[3] = (unsigned)((ga >> 32) & 0x01FFFFFFu)     // global_addr[56:32]
            | 0x80000000u;                           // type=2 (image)
    u32x8 g1;
    g1[0] = 0x00010000u;                             // wg_mask=0, data_size=2B
    g1[1] = ((unsigned)K & 0xFFFFu) << 16;           // tensor_dim0[15:0]
    g1[2] = (((unsigned)K >> 16) & 0xFFFFu)          // tensor_dim0[31:16]
            | (((unsigned)tileN & 0xFFFFu) << 16);   // tensor_dim1[15:0]
    g1[3] = ((unsigned)tileK & 0xFFFFu) << 16;       // tile_dim0 (K elems)
    g1[4] = (unsigned)tileN & 0xFFFFu;               // tile_dim1 (rows), tile_dim2=0
    g1[5] = (unsigned)K;                             // tensor_dim0_stride[31:0]
    g1[6] = 0u;
    g1[7] = 0u;
    asm volatile("tensor_load_to_lds %0, %1" :: "s"(g0), "s"(g1) : "memory");
}

// ---------------------------------------------------------------------------
// Weight transpose + bf16 convert:  W[K,N] fp32  ->  Wt[N,K] bf16
// ---------------------------------------------------------------------------
__global__ void wt_transpose(const float* __restrict__ W, u16* __restrict__ Wt,
                             int K, int N) {
    long idx = (long)blockIdx.x * 256 + threadIdx.x;
    long total = (long)K * N;
    if (idx >= total) return;
    int k = (int)(idx / N);
    int n = (int)(idx % N);
    Wt[(long)n * K + k] = f32_to_bf16(W[idx]);
}

// ---------------------------------------------------------------------------
// Build X = concat(word, entity) fp32 + bf16 copy
// ---------------------------------------------------------------------------
__global__ void concat_init(const float* __restrict__ wh, const float* __restrict__ eh,
                            float* __restrict__ X, u16* __restrict__ Xb) {
    long idx = (long)blockIdx.x * 256 + threadIdx.x;
    long total = (long)BB * S_TOT * H;
    if (idx >= total) return;
    int hh = (int)(idx % H);
    long r = idx / H;
    int s = (int)(r % S_TOT);
    int b = (int)(r / S_TOT);
    float v = (s < LW) ? wh[((long)b * LW + s) * H + hh]
                       : eh[((long)b * LE + (s - LW)) * H + hh];
    X[idx] = v;
    Xb[idx] = f32_to_bf16(v);
}

// ---------------------------------------------------------------------------
// Generic bf16 GEMM:  out[M,N] = A[M,K](bf16,row-major) * Bt[N,K]^T + bias
// block = 128 threads (4 waves); block tile 64(M) x 64(N);
// wave w: rows 16w..16w+15, cols 0..63 (4 accumulators of 16x16).
// B tiles (64 rows x 32 k) are TDM-loaded into per-wave double-buffered LDS,
// software-pipelined against the WMMA math (s_wait_tensorcnt).
// Epilogue: +bias, +residual (fp32), optional GELU, fp32 and/or bf16 stores,
// optional per-head transposed bf16 store (for V -> Vt[b][h][dh][s]).
// ---------------------------------------------------------------------------
__global__ void gemm_bf16(const u16* __restrict__ A, int lda,
                          const u16* __restrict__ Bt, int K,
                          const float* __restrict__ bias,
                          float* __restrict__ outF, u16* __restrict__ outB, int ldo,
                          const float* __restrict__ residual,
                          u16* __restrict__ vtOut,
                          int rowOff0, int rowStrideZ,
                          int gelu) {
    __shared__ u16 tileB[4][2][64 * 32];   // per-wave double-buffered B tile (32 KB)

    const int lane = threadIdx.x & 31;
    const int wave = threadIdx.x >> 5;
    const long rowBase = (long)blockIdx.z * rowStrideZ + rowOff0 +
                         (long)blockIdx.y * 64 + wave * 16;
    const int colBase = blockIdx.x * 64;
    const int kHalf = (lane < 16) ? 0 : 8;
    const int lRow  = lane & 15;

    unsigned ldsAddr0 = __builtin_amdgcn_readfirstlane(
        (unsigned)(unsigned long long)(uintptr_t)&tileB[wave][0][0]);
    unsigned ldsAddr1 = __builtin_amdgcn_readfirstlane(
        (unsigned)(unsigned long long)(uintptr_t)&tileB[wave][1][0]);

    v8f acc[4];
    #pragma unroll
    for (int j = 0; j < 4; ++j) acc[j] = (v8f)(0.f);

    const u16* aRow   = A + (rowBase + lRow) * (long)lda;
    const u16* btBase = Bt + (long)colBase * K;

    // prologue: TDM tile 0 into buffer 0, A fragment 0 into registers
    tdm_load_tile_bf16(btBase, ldsAddr0, K, 32, 64);
    Frag a_cur;
    a_cur.q[0] = *(const uint4*)(aRow + kHalf);
    a_cur.q[1] = *(const uint4*)(aRow + kHalf + 16);

    int bufSel = 0;
    for (int k0 = 0; k0 < K; k0 += 32) {
        const bool hasNext = (k0 + 32) < K;
        Frag a_next;
        if (hasNext) {
            // issue next B tile into the other buffer; prefetch + load next A frag
            tdm_load_tile_bf16(btBase + k0 + 32, bufSel ? ldsAddr0 : ldsAddr1,
                               K, 32, 64);
            __builtin_prefetch((const void*)(aRow + k0 + 32), 0, 1);
            a_next.q[0] = *(const uint4*)(aRow + k0 + 32 + kHalf);
            a_next.q[1] = *(const uint4*)(aRow + k0 + 32 + kHalf + 16);
        }
        // wait until the CURRENT tile's TDM is complete (TENSORcnt is in-order)
        if (hasNext) __builtin_amdgcn_s_wait_tensorcnt(1);
        else         __builtin_amdgcn_s_wait_tensorcnt(0);
        asm volatile("" ::: "memory");

        const u16* tb = &tileB[wave][bufSel][0];
        Frag bf[4];
        #pragma unroll
        for (int j = 0; j < 4; ++j) {
            const u16* bRow = tb + (j * 16 + lRow) * 32 + kHalf;
            bf[j].q[0] = *(const uint4*)(bRow);
            bf[j].q[1] = *(const uint4*)(bRow + 16);
        }
        #pragma unroll
        for (int j = 0; j < 4; ++j)
            acc[j] = __builtin_amdgcn_wmma_f32_16x16x32_bf16(
                false, a_cur.v, false, bf[j].v, (short)0, acc[j], false, false);

        if (hasNext) a_cur = a_next;
        bufSel ^= 1;
    }

    const int nLane = lane & 15;
    const int mBase = (lane >> 4) * 8;
    #pragma unroll
    for (int j = 0; j < 4; ++j) {
        int col = colBase + j * 16 + nLane;
        float bv = bias ? bias[col] : 0.f;
        #pragma unroll
        for (int r = 0; r < 8; ++r) {
            long row = rowBase + mBase + r;
            float x = acc[j][r] + bv;
            if (residual) x += residual[row * (long)ldo + col];
            if (gelu) x = 0.5f * x * (1.f + erff(x * 0.70710678118654752f));
            if (outF) outF[row * (long)ldo + col] = x;
            if (outB) outB[row * (long)ldo + col] = f32_to_bf16(x);
            if (vtOut) {
                int b = (int)(row / S_TOT);
                int s = (int)(row % S_TOT);
                int hh = col >> 6;
                int dh = col & 63;
                vtOut[(((long)b * NH + hh) * DH + dh) * S_TOT + s] = f32_to_bf16(x);
            }
        }
    }
}

// ---------------------------------------------------------------------------
// Entity-aware attention for one (batch, head, 16-query tile).
// block = 128 threads (4 waves).
// scores (fp32, 16 x S) in dynamic LDS; probs (bf16) in static LDS.
// Qw used against word keys, Qe against entity keys (both full-S buffers).
// ---------------------------------------------------------------------------
__global__ void attention_kernel(const u16* __restrict__ Qw, const u16* __restrict__ Qe,
                                 const u16* __restrict__ Kb, const u16* __restrict__ Vt,
                                 const float* __restrict__ mask,
                                 u16* __restrict__ ctxOut) {
    extern __shared__ float scores[];                 // [16][LDSROW] fp32
    __shared__ u16 probs[16 * LDSROW];                // [16][LDSROW] bf16
    __shared__ float redmax[16 * 8];
    __shared__ float redsum[16 * 8];

    const int qt = blockIdx.x;       // query 16-tile, 0..67
    const int h  = blockIdx.y;       // head
    const int b  = blockIdx.z;       // batch
    const int lane = threadIdx.x & 31;
    const int wave = threadIdx.x >> 5;
    const int kHalf = (lane < 16) ? 0 : 8;
    const int lRow  = lane & 15;
    const long qRow = (long)b * S_TOT + qt * 16;

    // Q fragments (word-directed and entity-directed), dh halves 0..31 / 32..63
    Frag aw0, aw1, ae0, ae1;
    {
        const u16* base = Qw + (qRow + lRow) * (long)H + h * DH;
        aw0.q[0] = *(const uint4*)(base + kHalf);
        aw0.q[1] = *(const uint4*)(base + kHalf + 16);
        aw1.q[0] = *(const uint4*)(base + 32 + kHalf);
        aw1.q[1] = *(const uint4*)(base + 32 + kHalf + 16);
        const u16* base2 = Qe + (qRow + lRow) * (long)H + h * DH;
        ae0.q[0] = *(const uint4*)(base2 + kHalf);
        ae0.q[1] = *(const uint4*)(base2 + kHalf + 16);
        ae1.q[0] = *(const uint4*)(base2 + 32 + kHalf);
        ae1.q[1] = *(const uint4*)(base2 + 32 + kHalf + 16);
    }

    // scores = Qsel . K^T  (per 16-key tile; K bf16 row-major is exactly the
    // transposed-B layout WMMA wants)
    const float scale = 0.125f;  // 1/sqrt(64)
    for (int kt = wave; kt < S_TOT / 16; kt += 4) {
        const bool isWord = (kt * 16) < LW;
        v8f c = (v8f)(0.f);
        const u16* kbase = Kb + ((long)b * S_TOT + kt * 16 + lRow) * H + h * DH;
        Frag b0, b1;
        b0.q[0] = *(const uint4*)(kbase + kHalf);
        b0.q[1] = *(const uint4*)(kbase + kHalf + 16);
        b1.q[0] = *(const uint4*)(kbase + 32 + kHalf);
        b1.q[1] = *(const uint4*)(kbase + 32 + kHalf + 16);
        if (isWord) {
            c = __builtin_amdgcn_wmma_f32_16x16x32_bf16(false, aw0.v, false, b0.v, (short)0, c, false, false);
            c = __builtin_amdgcn_wmma_f32_16x16x32_bf16(false, aw1.v, false, b1.v, (short)0, c, false, false);
        } else {
            c = __builtin_amdgcn_wmma_f32_16x16x32_bf16(false, ae0.v, false, b0.v, (short)0, c, false, false);
            c = __builtin_amdgcn_wmma_f32_16x16x32_bf16(false, ae1.v, false, b1.v, (short)0, c, false, false);
        }
        const int nLane = lane & 15;
        const int mBase = (lane >> 4) * 8;
        const int key = kt * 16 + nLane;
        const float mv = mask[(long)b * S_TOT + key];
        #pragma unroll
        for (int r = 0; r < 8; ++r)
            scores[(mBase + r) * LDSROW + key] = c[r] * scale + mv;
    }
    __syncthreads();

    // softmax over the full S=1088 keys; each row owned by 8 threads
    const int q   = threadIdx.x >> 3;   // 0..15
    const int seg = threadIdx.x & 7;    // 0..7
    const int segLen = S_TOT / 8;       // 136
    float m = -3.0e38f;
    for (int k = seg * segLen; k < (seg + 1) * segLen; ++k)
        m = fmaxf(m, scores[q * LDSROW + k]);
    redmax[q * 8 + seg] = m;
    __syncthreads();
    float rowm = -3.0e38f;
    #pragma unroll
    for (int i = 0; i < 8; ++i) rowm = fmaxf(rowm, redmax[q * 8 + i]);
    float ssum = 0.f;
    for (int k = seg * segLen; k < (seg + 1) * segLen; ++k) {
        float e = expf(scores[q * LDSROW + k] - rowm);
        scores[q * LDSROW + k] = e;
        ssum += e;
    }
    redsum[q * 8 + seg] = ssum;
    __syncthreads();
    float rowsum = 0.f;
    #pragma unroll
    for (int i = 0; i < 8; ++i) rowsum += redsum[q * 8 + i];
    const float inv = 1.f / rowsum;
    for (int k = seg * segLen; k < (seg + 1) * segLen; ++k)
        probs[q * LDSROW + k] = f32_to_bf16(scores[q * LDSROW + k] * inv);
    __syncthreads();

    // ctx = probs(16 x S) . V(S x 64); wave w handles dh tile w*16..w*16+15.
    // B-operand read from Vt[b][h][dh][key] (contiguous in key).
    v8f c = (v8f)(0.f);
    for (int ks = 0; ks < S_TOT; ks += 32) {
        Frag a;
        a.q[0] = *(const uint4*)(probs + lRow * LDSROW + ks + kHalf);
        a.q[1] = *(const uint4*)(probs + lRow * LDSROW + ks + kHalf + 16);
        const u16* vbase = Vt + (((long)b * NH + h) * DH + wave * 16 + lRow) * S_TOT + ks;
        Frag bf;
        bf.q[0] = *(const uint4*)(vbase + kHalf);
        bf.q[1] = *(const uint4*)(vbase + kHalf + 16);
        c = __builtin_amdgcn_wmma_f32_16x16x32_bf16(false, a.v, false, bf.v, (short)0, c, false, false);
    }
    const int nLane = lane & 15;
    const int mBase = (lane >> 4) * 8;
    #pragma unroll
    for (int r = 0; r < 8; ++r)
        ctxOut[(qRow + mBase + r) * H + h * DH + wave * 16 + nLane] = f32_to_bf16(c[r]);
}

// ---------------------------------------------------------------------------
// LayerNorm over H=768; one block (256 threads) per row.
// Writes fp32, bf16, and optionally a second fp32 copy (layer output states).
// ---------------------------------------------------------------------------
__global__ void layernorm_kernel(const float* __restrict__ in,
                                 const float* __restrict__ g, const float* __restrict__ bta,
                                 float* __restrict__ outF, u16* __restrict__ outB,
                                 float* __restrict__ outCopy) {
    const long row = blockIdx.x;
    const float* x = in + row * (long)H;
    __shared__ float buf[256];
    const int t = threadIdx.x;

    float s = 0.f;
    for (int i = t; i < H; i += 256) s += x[i];
    buf[t] = s;
    __syncthreads();
    for (int off = 128; off > 0; off >>= 1) {
        if (t < off) buf[t] += buf[t + off];
        __syncthreads();
    }
    const float mean = buf[0] * (1.0f / H);
    __syncthreads();

    float v = 0.f;
    for (int i = t; i < H; i += 256) {
        float d = x[i] - mean;
        v += d * d;
    }
    buf[t] = v;
    __syncthreads();
    for (int off = 128; off > 0; off >>= 1) {
        if (t < off) buf[t] += buf[t + off];
        __syncthreads();
    }
    const float invs = rsqrtf(buf[0] * (1.0f / H) + 1e-12f);

    for (int i = t; i < H; i += 256) {
        float y = (x[i] - mean) * invs * g[i] + bta[i];
        outF[row * (long)H + i] = y;
        outB[row * (long)H + i] = f32_to_bf16(y);
        if (outCopy) outCopy[row * (long)H + i] = y;
    }
}

// ---------------------------------------------------------------------------
// Host orchestration
// ---------------------------------------------------------------------------
extern "C" void kernel_launch(void* const* d_in, const int* in_sizes, int n_in,
                              void* d_out, int out_size, void* d_ws, size_t ws_size,
                              hipStream_t stream) {
    (void)in_sizes; (void)n_in; (void)out_size; (void)ws_size;

    const float* wh   = (const float*)d_in[0];
    const float* eh   = (const float*)d_in[1];
    const float* mask = (const float*)d_in[2];
    const float* Wq     = (const float*)d_in[3];  const float* bq     = (const float*)d_in[4];
    const float* Wk     = (const float*)d_in[5];  const float* bk     = (const float*)d_in[6];
    const float* Wv     = (const float*)d_in[7];  const float* bv     = (const float*)d_in[8];
    const float* Wq_w2e = (const float*)d_in[9];  const float* bq_w2e = (const float*)d_in[10];
    const float* Wq_e2w = (const float*)d_in[11]; const float* bq_e2w = (const float*)d_in[12];
    const float* Wq_e2e = (const float*)d_in[13]; const float* bq_e2e = (const float*)d_in[14];
    const float* Wo     = (const float*)d_in[15]; const float* bo     = (const float*)d_in[16];
    const float* Wi     = (const float*)d_in[17]; const float* bi     = (const float*)d_in[18];
    const float* Wo2    = (const float*)d_in[19]; const float* bo2    = (const float*)d_in[20];
    const float* ln1_g  = (const float*)d_in[21]; const float* ln1_b  = (const float*)d_in[22];
    const float* ln2_g  = (const float*)d_in[23]; const float* ln2_b  = (const float*)d_in[24];

    // workspace layout
    char* p = (char*)d_ws;
    auto alloc = [&](size_t bytes) -> void* {
        void* r = (void*)p;
        p += (bytes + 255) & ~(size_t)255;
        return r;
    };
    const long HH = (long)H * H;
    u16* WqT    = (u16*)alloc(LAYERS * HH * 2);
    u16* WkT    = (u16*)alloc(LAYERS * HH * 2);
    u16* WvT    = (u16*)alloc(LAYERS * HH * 2);
    u16* Ww2eT  = (u16*)alloc(LAYERS * HH * 2);
    u16* We2wT  = (u16*)alloc(LAYERS * HH * 2);
    u16* We2eT  = (u16*)alloc(LAYERS * HH * 2);
    u16* WoT    = (u16*)alloc(LAYERS * HH * 2);
    u16* WiT    = (u16*)alloc((long)LAYERS * H * FF * 2);
    u16* Wo2T   = (u16*)alloc((long)LAYERS * FF * H * 2);

    const long NTOK = (long)BB * S_TOT;
    float* X   = (float*)alloc(NTOK * H * 4);
    u16*   Xb  = (u16*)  alloc(NTOK * H * 2);
    u16*   Kb  = (u16*)  alloc(NTOK * H * 2);
    u16*   Qwb = (u16*)  alloc(NTOK * H * 2);
    u16*   Qeb = (u16*)  alloc(NTOK * H * 2);
    u16*   Vtb = (u16*)  alloc(NTOK * H * 2);
    u16*   Ctx = (u16*)  alloc(NTOK * H * 2);
    float* Y   = (float*)alloc(NTOK * H * 4);
    float* X2  = (float*)alloc(NTOK * H * 4);
    u16*   X2b = (u16*)  alloc(NTOK * H * 2);
    u16*   Ib  = (u16*)  alloc(NTOK * FF * 2);

    // weight conversion (transpose + bf16), once per launch
    const int gHH = (int)((HH + 255) / 256);
    const int gHF = (int)(((long)H * FF + 255) / 256);
    for (int l = 0; l < LAYERS; ++l) {
        wt_transpose<<<gHH, 256, 0, stream>>>(Wq     + l * HH, WqT   + l * HH, H, H);
        wt_transpose<<<gHH, 256, 0, stream>>>(Wk     + l * HH, WkT   + l * HH, H, H);
        wt_transpose<<<gHH, 256, 0, stream>>>(Wv     + l * HH, WvT   + l * HH, H, H);
        wt_transpose<<<gHH, 256, 0, stream>>>(Wq_w2e + l * HH, Ww2eT + l * HH, H, H);
        wt_transpose<<<gHH, 256, 0, stream>>>(Wq_e2w + l * HH, We2wT + l * HH, H, H);
        wt_transpose<<<gHH, 256, 0, stream>>>(Wq_e2e + l * HH, We2eT + l * HH, H, H);
        wt_transpose<<<gHH, 256, 0, stream>>>(Wo     + l * HH, WoT   + l * HH, H, H);
        wt_transpose<<<gHF, 256, 0, stream>>>(Wi  + (long)l * H * FF, WiT  + (long)l * H * FF, H, FF);
        wt_transpose<<<gHF, 256, 0, stream>>>(Wo2 + (long)l * FF * H, Wo2T + (long)l * FF * H, FF, H);
    }

    // initial X = concat(word, entity)
    {
        long tot = NTOK * H;
        concat_init<<<(int)((tot + 255) / 256), 256, 0, stream>>>(wh, eh, X, Xb);
    }

    float* out = (float*)d_out;
    const dim3 full(H / 64, (int)(NTOK / 64), 1);     // 12 x 34
    const dim3 ffn(FF / 64, (int)(NTOK / 64), 1);     // 48 x 34
    const dim3 qword(H / 64, LW / 64, BB);            // 12 x 16 x 2
    const dim3 qent(H / 64, LE / 64, BB);             // 12 x 1 x 2

    for (int l = 0; l < LAYERS; ++l) {
        const u16* WkTl = WkT + l * HH;   const float* bkl = bk + l * H;
        const u16* WvTl = WvT + l * HH;   const float* bvl = bv + l * H;
        const u16* WqTl = WqT + l * HH;   const float* bql = bq + l * H;
        const u16* Ww2eTl = Ww2eT + l * HH; const float* bw2el = bq_w2e + l * H;
        const u16* We2wTl = We2wT + l * HH; const float* be2wl = bq_e2w + l * H;
        const u16* We2eTl = We2eT + l * HH; const float* be2el = bq_e2e + l * H;
        const u16* WoTl = WoT + l * HH;   const float* bol = bo + l * H;
        const u16* WiTl = WiT + (long)l * H * FF;   const float* bil = bi + (long)l * FF;
        const u16* Wo2Tl = Wo2T + (long)l * FF * H; const float* bo2l = bo2 + l * H;

        // K, V projections over all tokens
        gemm_bf16<<<full, 128, 0, stream>>>(Xb, H, WkTl, H, bkl,
                                            nullptr, Kb, H, nullptr, nullptr, 0, 0, 0);
        gemm_bf16<<<full, 128, 0, stream>>>(Xb, H, WvTl, H, bvl,
                                            nullptr, nullptr, H, nullptr, Vtb, 0, 0, 0);
        // entity-aware Q: word rows then entity rows into Qw / Qe
        gemm_bf16<<<qword, 128, 0, stream>>>(Xb, H, WqTl, H, bql,
                                             nullptr, Qwb, H, nullptr, nullptr, 0, S_TOT, 0);
        gemm_bf16<<<qent, 128, 0, stream>>>(Xb, H, We2wTl, H, be2wl,
                                            nullptr, Qwb, H, nullptr, nullptr, LW, S_TOT, 0);
        gemm_bf16<<<qword, 128, 0, stream>>>(Xb, H, Ww2eTl, H, bw2el,
                                             nullptr, Qeb, H, nullptr, nullptr, 0, S_TOT, 0);
        gemm_bf16<<<qent, 128, 0, stream>>>(Xb, H, We2eTl, H, be2el,
                                            nullptr, Qeb, H, nullptr, nullptr, LW, S_TOT, 0);

        // attention (scores LDS = 16 * LDSROW fp32, dynamic)
        attention_kernel<<<dim3(S_TOT / 16, NH, BB), 128,
                           16 * LDSROW * sizeof(float), stream>>>(
            Qwb, Qeb, Kb, Vtb, mask, Ctx);

        // output projection + residual -> Y ; LN1 -> X2 (fp32 + bf16)
        gemm_bf16<<<full, 128, 0, stream>>>(Ctx, H, WoTl, H, bol,
                                            Y, nullptr, H, X, nullptr, 0, 0, 0);
        layernorm_kernel<<<(int)NTOK, 256, 0, stream>>>(Y, ln1_g + l * H, ln1_b + l * H,
                                                        X2, X2b, nullptr);

        // FFN: GELU(X2 @ Wi + bi) @ Wo2 + bo2 + X2 ; LN2 -> X (next layer) + states[l]
        gemm_bf16<<<ffn, 128, 0, stream>>>(X2b, H, WiTl, H, bil,
                                           nullptr, Ib, FF, nullptr, nullptr, 0, 0, 1);
        gemm_bf16<<<full, 128, 0, stream>>>(Ib, FF, Wo2Tl, FF, bo2l,
                                            Y, nullptr, H, X2, nullptr, 0, 0, 0);
        layernorm_kernel<<<(int)NTOK, 256, 0, stream>>>(Y, ln2_g + l * H, ln2_b + l * H,
                                                        X, Xb, out + (long)l * NTOK * H);
    }
}
